// AxialTL_12506944766487
// MI455X (gfx1250) — matrix operand order repaired and checked by reading, not compiled, not run on hardware
//
#include <hip/hip_runtime.h>
#include <hip/hip_bf16.h>
#include <cmath>

#define E_DIM 768
#define F_DIM 3072
#define H_DIM 12
#define D_DIM 64
#define R_DIM 128
#define C_DIM 384
#define M_TOT (R_DIM * C_DIM)
#define LN_EPS 1e-5f

typedef __bf16 bf16_t;
typedef __attribute__((ext_vector_type(16))) __bf16 v16bf;
typedef __attribute__((ext_vector_type(8)))  __bf16 v8bf;
typedef __attribute__((ext_vector_type(8)))  float  v8f;

// float -> bf16, round-to-nearest-even
__device__ __forceinline__ bf16_t f2bf(float f) {
  union { float f; unsigned u; } in; in.f = f;
  unsigned u = in.u;
  unsigned r = (u + 0x7fffu + ((u >> 16) & 1u)) >> 16;
  union { unsigned short s; bf16_t b; } out; out.s = (unsigned short)r;
  return out.b;
}

__device__ __forceinline__ v8f wmma_bf16(v16bf a, v16bf b, v8f c) {
  return __builtin_amdgcn_wmma_f32_16x16x32_bf16(false, a, false, b, (short)0, c,
                                                 false, false);
}

// A fragment 16x32 bf16: lane l<16 -> row m=l, K = {0..7, 16..23};
// lane l>=16 -> row m=l-16, K = {8..15, 24..31}.
__device__ __forceinline__ v16bf load_a_frag(const bf16_t* base, long lda, int lane) {
  int m  = lane & 15;
  int ko = (lane >> 4) * 8;
  const bf16_t* p = base + (long)m * lda + ko;
  v8bf lo = *(const v8bf*)(p);
  v8bf hi = *(const v8bf*)(p + 16);
  return __builtin_shufflevector(lo, hi, 0,1,2,3,4,5,6,7,8,9,10,11,12,13,14,15);
}

// B fragment 32x16 bf16 (operand element (k,n) = base[n*ldb + k], K contiguous):
// lane l<16 -> col n=l, K = {0..7, 16..23}; lane l>=16 -> col n=l-16, K = {8..15, 24..31}.
__device__ __forceinline__ v16bf load_b_frag(const bf16_t* base, long ldb, int lane) {
  int n  = lane & 15;
  int ko = (lane >> 4) * 8;
  const bf16_t* p = base + (long)n * ldb + ko;
  v8bf lo = *(const v8bf*)(p);
  v8bf hi = *(const v8bf*)(p + 16);
  return __builtin_shufflevector(lo, hi, 0,1,2,3,4,5,6,7,8,9,10,11,12,13,14,15);
}

// ---------------------------------------------------------------------------
__global__ void k_f32_to_bf16(const float* __restrict__ s, bf16_t* __restrict__ d, int n) {
  int i = blockIdx.x * 256 + threadIdx.x;
  if (i < n) d[i] = f2bf(s[i]);
}

// LayerNorm over E=768, one block (256 thr) per token, bf16 output
__global__ void __launch_bounds__(256) k_layernorm(const float* __restrict__ x,
                                                   const float* __restrict__ g,
                                                   const float* __restrict__ b,
                                                   bf16_t* __restrict__ out) {
  long t = blockIdx.x;
  const float* xp = x + t * E_DIM;
  bf16_t* op = out + t * E_DIM;
  int tid = threadIdx.x;
  __shared__ float red[256];
  float a0 = xp[tid], a1 = xp[tid + 256], a2 = xp[tid + 512];
  red[tid] = a0 + a1 + a2; __syncthreads();
  for (int o = 128; o > 0; o >>= 1) { if (tid < o) red[tid] += red[tid + o]; __syncthreads(); }
  float mean = red[0] * (1.0f / E_DIM); __syncthreads();
  float d0 = a0 - mean, d1 = a1 - mean, d2 = a2 - mean;
  red[tid] = d0 * d0 + d1 * d1 + d2 * d2; __syncthreads();
  for (int o = 128; o > 0; o >>= 1) { if (tid < o) red[tid] += red[tid + o]; __syncthreads(); }
  float rstd = rsqrtf(red[0] * (1.0f / E_DIM) + LN_EPS);
  op[tid]       = f2bf(d0 * rstd * g[tid]       + b[tid]);
  op[tid + 256] = f2bf(d1 * rstd * g[tid + 256] + b[tid + 256]);
  op[tid + 512] = f2bf(d2 * rstd * g[tid + 512] + b[tid + 512]);
}

// GEMM: out[m,n] = epilogue( sum_k A[m,k]*W[n,k] + bias[n] )
// block = 256 thr = 8 waves (2 in M x 4 in N); wave tile 64x32 -> block tile 128x128
// MODE 0: f32 out = res + v
// MODE 1: bf16 out = v*scale
// MODE 2: bf16 out = gelu(v)
// MODE 3: bf16 out = v, stored transposed for row-attn V: Vt[((r*H+h)*D+d)*C + i]
// MODE 4: bf16 out = v, stored transposed for col-attn V: Vt[((c*H+h)*D+d)*R + i]
template <int MODE>
__global__ void __launch_bounds__(256) k_gemm(const bf16_t* __restrict__ A,
                                              const bf16_t* __restrict__ W,
                                              const float* __restrict__ bias, float scale,
                                              const float* __restrict__ res,
                                              float* __restrict__ outf,
                                              bf16_t* __restrict__ outb,
                                              int M, int N, int K) {
  (void)M;
  int lane = threadIdx.x & 31;
  int wave = threadIdx.x >> 5;
  int wm = wave & 1, wn = wave >> 1;
  int m0 = blockIdx.x * 128 + wm * 64;
  int n0 = blockIdx.y * 128 + wn * 32;
  v8f acc[4][2] = {};
  for (int k0 = 0; k0 < K; k0 += 32) {
    v16bf a0 = load_a_frag(A + (long)(m0     ) * K + k0, K, lane);
    v16bf a1 = load_a_frag(A + (long)(m0 + 16) * K + k0, K, lane);
    v16bf a2 = load_a_frag(A + (long)(m0 + 32) * K + k0, K, lane);
    v16bf a3 = load_a_frag(A + (long)(m0 + 48) * K + k0, K, lane);
    v16bf b0 = load_b_frag(W + (long)n0 * K + k0, K, lane);
    v16bf b1 = load_b_frag(W + (long)(n0 + 16) * K + k0, K, lane);
    acc[0][0] = wmma_bf16(a0, b0, acc[0][0]);
    acc[0][1] = wmma_bf16(a0, b1, acc[0][1]);
    acc[1][0] = wmma_bf16(a1, b0, acc[1][0]);
    acc[1][1] = wmma_bf16(a1, b1, acc[1][1]);
    acc[2][0] = wmma_bf16(a2, b0, acc[2][0]);
    acc[2][1] = wmma_bf16(a2, b1, acc[2][1]);
    acc[3][0] = wmma_bf16(a3, b0, acc[3][0]);
    acc[3][1] = wmma_bf16(a3, b1, acc[3][1]);
  }
  int rl = (lane >> 4) * 8, cl = lane & 15;
#pragma unroll
  for (int sm = 0; sm < 4; ++sm)
#pragma unroll
    for (int sn = 0; sn < 2; ++sn) {
      int col = n0 + sn * 16 + cl;
      float bv = bias[col];
#pragma unroll
      for (int ii = 0; ii < 8; ++ii) {
        int row = m0 + sm * 16 + ii + rl;
        float v = acc[sm][sn][ii] + bv;
        if (MODE == 0) {
          long idx = (long)row * N + col;
          outf[idx] = res[idx] + v;
        } else if (MODE == 1) {
          outb[(long)row * N + col] = f2bf(v * scale);
        } else if (MODE == 2) {
          outb[(long)row * N + col] =
              f2bf(0.5f * v * (1.0f + erff(v * 0.70710678118654752f)));
        } else if (MODE == 3) {
          int r = row / C_DIM, i = row % C_DIM;
          int h = col / D_DIM, d = col % D_DIM;
          outb[(((long)r * H_DIM + h) * D_DIM + d) * C_DIM + i] = f2bf(v);
        } else {  // MODE 4
          int i = row / C_DIM, c = row % C_DIM;
          int h = col / D_DIM, d = col % D_DIM;
          outb[(((long)c * H_DIM + h) * D_DIM + d) * R_DIM + i] = f2bf(v);
        }
      }
    }
}

// Row-attention logits: L[h,i,j] = sum_{r,d} Q[r,i,h,d]*K[r,j,h,d]
// grid (C/128, C/64, H), block 256 (8 waves, 4x2), wave 32x32
__global__ void __launch_bounds__(256) k_row_logits(const bf16_t* __restrict__ Q,
                                                    const bf16_t* __restrict__ Kp,
                                                    float* __restrict__ L) {
  int lane = threadIdx.x & 31;
  int wave = threadIdx.x >> 5;
  int wm = wave & 3, wn = wave >> 2;
  int h  = blockIdx.z;
  int i0 = blockIdx.x * 128 + wm * 32;
  int j0 = blockIdx.y * 64 + wn * 32;
  v8f acc[2][2] = {};
  for (int r = 0; r < R_DIM; ++r) {
    const bf16_t* qb = Q  + (long)r * C_DIM * E_DIM + h * D_DIM;
    const bf16_t* kb = Kp + (long)r * C_DIM * E_DIM + h * D_DIM;
#pragma unroll
    for (int dc = 0; dc < D_DIM; dc += 32) {
      v16bf a0 = load_a_frag(qb + (long)i0 * E_DIM + dc, E_DIM, lane);
      v16bf a1 = load_a_frag(qb + (long)(i0 + 16) * E_DIM + dc, E_DIM, lane);
      v16bf b0 = load_b_frag(kb + (long)j0 * E_DIM + dc, E_DIM, lane);
      v16bf b1 = load_b_frag(kb + (long)(j0 + 16) * E_DIM + dc, E_DIM, lane);
      acc[0][0] = wmma_bf16(a0, b0, acc[0][0]);
      acc[0][1] = wmma_bf16(a0, b1, acc[0][1]);
      acc[1][0] = wmma_bf16(a1, b0, acc[1][0]);
      acc[1][1] = wmma_bf16(a1, b1, acc[1][1]);
    }
  }
  int rl = (lane >> 4) * 8, cl = lane & 15;
#pragma unroll
  for (int sm = 0; sm < 2; ++sm)
#pragma unroll
    for (int sn = 0; sn < 2; ++sn)
#pragma unroll
      for (int ii = 0; ii < 8; ++ii) {
        int i = i0 + sm * 16 + ii + rl;
        int j = j0 + sn * 16 + cl;
        L[((long)h * C_DIM + i) * C_DIM + j] = acc[sm][sn][ii];
      }
}

// softmax over last dim (C=384) of L[H*C, C] -> bf16 P
__global__ void __launch_bounds__(128) k_softmax_rows(const float* __restrict__ L,
                                                      bf16_t* __restrict__ P) {
  long row = blockIdx.x;
  const float* lp = L + row * C_DIM;
  bf16_t* pp = P + row * C_DIM;
  int tid = threadIdx.x;
  __shared__ float red[128];
  float v0 = lp[tid], v1 = lp[tid + 128], v2 = lp[tid + 256];
  float m = fmaxf(v0, fmaxf(v1, v2));
  red[tid] = m; __syncthreads();
  for (int o = 64; o > 0; o >>= 1) { if (tid < o) red[tid] = fmaxf(red[tid], red[tid + o]); __syncthreads(); }
  m = red[0]; __syncthreads();
  float e0 = __expf(v0 - m), e1 = __expf(v1 - m), e2 = __expf(v2 - m);
  red[tid] = e0 + e1 + e2; __syncthreads();
  for (int o = 64; o > 0; o >>= 1) { if (tid < o) red[tid] += red[tid + o]; __syncthreads(); }
  float inv = 1.0f / red[0];
  pp[tid] = f2bf(e0 * inv); pp[tid + 128] = f2bf(e1 * inv); pp[tid + 256] = f2bf(e2 * inv);
}

// ctx[r,i,h,d] = sum_j P[h,i,j] * Vt[r,h,d,j]   (V pre-transposed per head)
// grid (C/128, R, H), block 128 (4 waves), wave: 32 rows x 64 cols(d)
__global__ void __launch_bounds__(128) k_row_ctx(const bf16_t* __restrict__ P,
                                                 const bf16_t* __restrict__ Vt,
                                                 bf16_t* __restrict__ ctx) {
  int lane = threadIdx.x & 31;
  int wave = threadIdx.x >> 5;
  int r = blockIdx.y, h = blockIdx.z;
  int i0 = blockIdx.x * 128 + wave * 32;
  const bf16_t* Ph = P + (long)h * C_DIM * C_DIM;
  const bf16_t* Vh = Vt + ((long)r * H_DIM + h) * D_DIM * C_DIM;  // [d][j], j contiguous
  v8f acc[2][4] = {};
  for (int jc = 0; jc < C_DIM; jc += 32) {
    v16bf a0 = load_a_frag(Ph + (long)i0 * C_DIM + jc, C_DIM, lane);
    v16bf a1 = load_a_frag(Ph + (long)(i0 + 16) * C_DIM + jc, C_DIM, lane);
#pragma unroll
    for (int sn = 0; sn < 4; ++sn) {
      v16bf bf = load_b_frag(Vh + (long)sn * 16 * C_DIM + jc, C_DIM, lane);
      acc[0][sn] = wmma_bf16(a0, bf, acc[0][sn]);
      acc[1][sn] = wmma_bf16(a1, bf, acc[1][sn]);
    }
  }
  int rl = (lane >> 4) * 8, cl = lane & 15;
#pragma unroll
  for (int sm = 0; sm < 2; ++sm)
#pragma unroll
    for (int sn = 0; sn < 4; ++sn)
#pragma unroll
      for (int ii = 0; ii < 8; ++ii) {
        int i = i0 + sm * 16 + ii + rl;
        ctx[((long)r * C_DIM + i) * E_DIM + h * D_DIM + sn * 16 + cl] = f2bf(acc[sm][sn][ii]);
      }
}

// Fused column attention, one block per (h,c): logits(128x128) -> softmax -> P@V
// block 256 = 8 waves; wave w owns query rows [16w,16w+16)
// Vt layout: Vt[((c*H+h)*D+d)*R + j]  (j contiguous)
__global__ void __launch_bounds__(256) k_col_attn(const bf16_t* __restrict__ Qp,
                                                  const bf16_t* __restrict__ Kp,
                                                  const bf16_t* __restrict__ Vt,
                                                  bf16_t* __restrict__ ctx) {
  int lane = threadIdx.x & 31;
  int w = threadIdx.x >> 5;
  int h = blockIdx.x, c = blockIdx.y;
  int i0 = w * 16;
  long ldrow = (long)C_DIM * E_DIM;  // stride between consecutive rows i
  const bf16_t* qb = Qp + ((long)i0 * C_DIM + c) * E_DIM + h * D_DIM;
  const bf16_t* kb = Kp + (long)c * E_DIM + h * D_DIM;
  const bf16_t* vb = Vt + ((long)c * H_DIM + h) * D_DIM * R_DIM;  // [d][j]
  // phase 1: logits, K-dim = d (64)
  v8f logit[8] = {};
#pragma unroll
  for (int dc = 0; dc < D_DIM; dc += 32) {
    v16bf a = load_a_frag(qb + dc, ldrow, lane);
#pragma unroll
    for (int jt = 0; jt < 8; ++jt) {
      v16bf bf = load_b_frag(kb + (long)jt * 16 * ldrow + dc, ldrow, lane);
      logit[jt] = wmma_bf16(a, bf, logit[jt]);
    }
  }
  // phase 2: row softmax (row = ii + 8*(lane>>4); row lives in one 16-lane half)
  __shared__ __align__(16) bf16_t Plds[8][16][128];
  int rl = (lane >> 4) * 8, cl = lane & 15;
#pragma unroll
  for (int ii = 0; ii < 8; ++ii) {
    float m = -1e30f;
#pragma unroll
    for (int jt = 0; jt < 8; ++jt) m = fmaxf(m, logit[jt][ii]);
    for (int o = 1; o < 16; o <<= 1) m = fmaxf(m, __shfl_xor(m, o, 32));
    float e[8]; float s = 0.f;
#pragma unroll
    for (int jt = 0; jt < 8; ++jt) { e[jt] = __expf(logit[jt][ii] - m); s += e[jt]; }
    for (int o = 1; o < 16; o <<= 1) s += __shfl_xor(s, o, 32);
    float inv = 1.0f / s;
#pragma unroll
    for (int jt = 0; jt < 8; ++jt) Plds[w][ii + rl][jt * 16 + cl] = f2bf(e[jt] * inv);
  }
  __syncthreads();
  // phase 3: ctx(16x64) = P(16x128) @ V(128x64), V loaded from transposed layout
  v8f acc[4] = {};
#pragma unroll
  for (int kc = 0; kc < 4; ++kc) {
    v16bf a = load_a_frag(&Plds[w][0][kc * 32], 128, lane);
#pragma unroll
    for (int sn = 0; sn < 4; ++sn) {
      v16bf bf = load_b_frag(vb + (long)sn * 16 * R_DIM + kc * 32, R_DIM, lane);
      acc[sn] = wmma_bf16(a, bf, acc[sn]);
    }
  }
#pragma unroll
  for (int sn = 0; sn < 4; ++sn)
#pragma unroll
    for (int ii = 0; ii < 8; ++ii) {
      int iq = i0 + ii + rl;
      ctx[((long)iq * C_DIM + c) * E_DIM + h * D_DIM + sn * 16 + cl] = f2bf(acc[sn][ii]);
    }
}

// ---------------------------------------------------------------------------
extern "C" void kernel_launch(void* const* d_in, const int* in_sizes, int n_in,
                              void* d_out, int out_size, void* d_ws, size_t ws_size,
                              hipStream_t stream) {
  (void)in_sizes; (void)n_in; (void)out_size; (void)ws_size;
  const float* x      = (const float*)d_in[0];
  const float* ln1_g  = (const float*)d_in[1];
  const float* ln1_b  = (const float*)d_in[2];
  const float* row_wq = (const float*)d_in[3];
  const float* row_bq = (const float*)d_in[4];
  const float* row_wk = (const float*)d_in[5];
  const float* row_bk = (const float*)d_in[6];
  const float* row_wv = (const float*)d_in[7];
  const float* row_bv = (const float*)d_in[8];
  const float* row_wo = (const float*)d_in[9];
  const float* row_bo = (const float*)d_in[10];
  const float* ln2_g  = (const float*)d_in[11];
  const float* ln2_b  = (const float*)d_in[12];
  const float* col_wq = (const float*)d_in[13];
  const float* col_bq = (const float*)d_in[14];
  const float* col_wk = (const float*)d_in[15];
  const float* col_bk = (const float*)d_in[16];
  const float* col_wv = (const float*)d_in[17];
  const float* col_bv = (const float*)d_in[18];
  const float* col_wo = (const float*)d_in[19];
  const float* col_bo = (const float*)d_in[20];
  const float* ln3_g  = (const float*)d_in[21];
  const float* ln3_b  = (const float*)d_in[22];
  const float* fc1_w  = (const float*)d_in[23];
  const float* fc1_b  = (const float*)d_in[24];
  const float* fc2_w  = (const float*)d_in[25];
  const float* fc2_b  = (const float*)d_in[26];
  float* out = (float*)d_out;

  char* ws = (char*)d_ws;
  size_t off = 0;
  auto alloc = [&](size_t bytes) -> void* {
    off = (off + 255) & ~(size_t)255;
    void* p = ws + off; off += bytes; return p;
  };
  const size_t EE = (size_t)E_DIM * E_DIM;
  const size_t FE = (size_t)F_DIM * E_DIM;
  const size_t ME = (size_t)M_TOT * E_DIM;
  bf16_t* wq_r = (bf16_t*)alloc(EE * 2);
  bf16_t* wk_r = (bf16_t*)alloc(EE * 2);
  bf16_t* wv_r = (bf16_t*)alloc(EE * 2);
  bf16_t* wo_r = (bf16_t*)alloc(EE * 2);
  bf16_t* wq_c = (bf16_t*)alloc(EE * 2);
  bf16_t* wk_c = (bf16_t*)alloc(EE * 2);
  bf16_t* wv_c = (bf16_t*)alloc(EE * 2);
  bf16_t* wo_c = (bf16_t*)alloc(EE * 2);
  bf16_t* w_f1 = (bf16_t*)alloc(FE * 2);
  bf16_t* w_f2 = (bf16_t*)alloc(FE * 2);
  bf16_t* Abuf = (bf16_t*)alloc(ME * 2);
  bf16_t* Qbuf = (bf16_t*)alloc(ME * 2);
  bf16_t* Kbuf = (bf16_t*)alloc(ME * 2);
  bf16_t* Vbuf = (bf16_t*)alloc(ME * 2);   // holds transposed V layouts
  float*  Lbuf = (float*)alloc((size_t)H_DIM * C_DIM * C_DIM * 4);
  bf16_t* Pbuf = (bf16_t*)alloc((size_t)H_DIM * C_DIM * C_DIM * 2);
  bf16_t* Hbuf = Qbuf;  // FFN hidden chunk (48MB) reuses Q region (75MB)

  auto cvt = [&](const float* s, bf16_t* d, size_t n) {
    k_f32_to_bf16<<<dim3((unsigned)((n + 255) / 256)), dim3(256), 0, stream>>>(s, d, (int)n);
  };
  cvt(row_wq, wq_r, EE); cvt(row_wk, wk_r, EE); cvt(row_wv, wv_r, EE); cvt(row_wo, wo_r, EE);
  cvt(col_wq, wq_c, EE); cvt(col_wk, wk_c, EE); cvt(col_wv, wv_c, EE); cvt(col_wo, wo_c, EE);
  cvt(fc1_w, w_f1, FE);  cvt(fc2_w, w_f2, FE);

  const dim3 blk(256);
  const dim3 gproj(M_TOT / 128, E_DIM / 128);
  const float qscale_row = 0.125f / sqrtf((float)R_DIM);  // d^-0.5 / sqrt(R)
  const float qscale_col = 0.125f;                        // d^-0.5

  // ---- row attention block ----
  k_layernorm<<<M_TOT, 256, 0, stream>>>(x, ln1_g, ln1_b, Abuf);
  k_gemm<1><<<gproj, blk, 0, stream>>>(Abuf, wq_r, row_bq, qscale_row, nullptr, nullptr, Qbuf,
                                       M_TOT, E_DIM, E_DIM);
  k_gemm<1><<<gproj, blk, 0, stream>>>(Abuf, wk_r, row_bk, 1.0f, nullptr, nullptr, Kbuf,
                                       M_TOT, E_DIM, E_DIM);
  k_gemm<3><<<gproj, blk, 0, stream>>>(Abuf, wv_r, row_bv, 1.0f, nullptr, nullptr, Vbuf,
                                       M_TOT, E_DIM, E_DIM);
  k_row_logits<<<dim3(C_DIM / 128, C_DIM / 64, H_DIM), blk, 0, stream>>>(Qbuf, Kbuf, Lbuf);
  k_softmax_rows<<<H_DIM * C_DIM, 128, 0, stream>>>(Lbuf, Pbuf);
  k_row_ctx<<<dim3(C_DIM / 128, R_DIM, H_DIM), 128, 0, stream>>>(Pbuf, Vbuf, Abuf);
  k_gemm<0><<<gproj, blk, 0, stream>>>(Abuf, wo_r, row_bo, 1.0f, x, out, nullptr,
                                       M_TOT, E_DIM, E_DIM);

  // ---- column attention block ----
  k_layernorm<<<M_TOT, 256, 0, stream>>>(out, ln2_g, ln2_b, Abuf);
  k_gemm<1><<<gproj, blk, 0, stream>>>(Abuf, wq_c, col_bq, qscale_col, nullptr, nullptr, Qbuf,
                                       M_TOT, E_DIM, E_DIM);
  k_gemm<1><<<gproj, blk, 0, stream>>>(Abuf, wk_c, col_bk, 1.0f, nullptr, nullptr, Kbuf,
                                       M_TOT, E_DIM, E_DIM);
  k_gemm<4><<<gproj, blk, 0, stream>>>(Abuf, wv_c, col_bv, 1.0f, nullptr, nullptr, Vbuf,
                                       M_TOT, E_DIM, E_DIM);
  k_col_attn<<<dim3(H_DIM, C_DIM), blk, 0, stream>>>(Qbuf, Kbuf, Vbuf, Abuf);
  k_gemm<0><<<gproj, blk, 0, stream>>>(Abuf, wo_c, col_bo, 1.0f, out, out, nullptr,
                                       M_TOT, E_DIM, E_DIM);

  // ---- FFN block (chunked over rows so hidden fits in workspace) ----
  k_layernorm<<<M_TOT, 256, 0, stream>>>(out, ln3_g, ln3_b, Abuf);
  const int CH = 8192;
  for (int c0 = 0; c0 < M_TOT; c0 += CH) {
    k_gemm<2><<<dim3(CH / 128, F_DIM / 128), blk, 0, stream>>>(
        Abuf + (size_t)c0 * E_DIM, w_f1, fc1_b, 1.0f, nullptr, nullptr, Hbuf,
        CH, F_DIM, E_DIM);
    k_gemm<0><<<dim3(CH / 128, E_DIM / 128), blk, 0, stream>>>(
        Hbuf, w_f2, fc2_b, 1.0f, out + (size_t)c0 * E_DIM, out + (size_t)c0 * E_DIM, nullptr,
        CH, E_DIM, F_DIM);
  }
}
